// CrossAttentionHead_14061722927720
// MI455X (gfx1250) — matrix operand order repaired and checked by reading, not compiled, not run on hardware
//
#include <hip/hip_runtime.h>
#include <hip/hip_bf16.h>

// ---------------------------------------------------------------------------
// CrossAttentionHead for MI455X (gfx1250): algebraically reduced attention.
// Heavy GEMMs (key projection, Q^T K energy) run on v_wmma_f32_16x16x32_bf16.
// B=64, C=512, H=W=28, N=784 (=49*16), KS=7.
//
// Reduction used (exact): final[b,c] = gamma/N * (value_w[c].y[b] +
//   value_b[c]*sum(s)) + mean_n x[b,c,n], where s[m] = sum_n softmax(QK)[n,m]
//   and y[b,ci] = sum_m s[m] * x[b,ci,m].  This eliminates the V projection
//   and the C x N x N output GEMM entirely (~66 of ~135 GFLOP).
// ---------------------------------------------------------------------------

typedef __bf16 bf16;
typedef __attribute__((ext_vector_type(16))) bf16  v16bf;
typedef __attribute__((ext_vector_type(8)))  float v8f;

#define BATCH 64
#define CH    512
#define NPOS  784      // 28*28, = 49 tiles of 16
#define NT    49       // position tiles

// Load a 32-byte WMMA bf16 fragment half-pair as two b128 global loads.
__device__ __forceinline__ v16bf load_frag(const bf16* p0, const bf16* p1) {
    union U { v16bf v; float4 f[2]; } u;
    u.f[0] = *reinterpret_cast<const float4*>(p0);
    u.f[1] = *reinterpret_cast<const float4*>(p1);
    return u.v;
}

// ---------------------------------------------------------------------------
// Kernel 0: key_w f32 -> bf16 (layout preserved: [co][ci], ci contiguous,
// which is exactly the B-operand-friendly layout for the K-projection GEMM).
// ---------------------------------------------------------------------------
__global__ void convert_w_kernel(const float* __restrict__ w, bf16* __restrict__ wb) {
    int i = blockIdx.x * 256 + threadIdx.x;
    if (i < CH * CH) wb[i] = (bf16)w[i];
}

// ---------------------------------------------------------------------------
// Kernel 1 (per batch): spatial attention (channel mean/max -> 7x7 conv ->
// sigmoid), then LDS-tiled transpose-pack of Xt[b][n][c] and
// Qt[b][n][c] = x*att as bf16 (transposed so WMMA fragments are contiguous).
// Transpose tile: 32 channels x 64 positions, padded row of 65 floats so both
// access phases are bank-conflict-free; global reads are 128B/wave coalesced,
// bf16 writes are 64B/wave contiguous.
// ---------------------------------------------------------------------------
__global__ void prep_kernel(const float* __restrict__ x,
                            const float* __restrict__ sa_w,
                            bf16* __restrict__ Xt, bf16* __restrict__ Qt) {
    __shared__ float sAvg[NPOS];
    __shared__ float sMax[NPOS];
    __shared__ float sAtt[NPOS];
    __shared__ float tile[32][65];
    const int b = blockIdx.x, t = threadIdx.x;
    const float* xb = x + (size_t)b * CH * NPOS;

    // channel mean/max per spatial position (lanes across positions: coalesced)
    for (int p = t; p < NPOS; p += 256) {
        float acc = 0.f, mx = -3.4e38f;
        for (int c = 0; c < CH; ++c) {
            float v = xb[(size_t)c * NPOS + p];
            acc += v; mx = fmaxf(mx, v);
        }
        sAvg[p] = acc * (1.0f / CH);
        sMax[p] = mx;
    }
    __syncthreads();

    // 7x7 SAME conv over [avg,max] channels + sigmoid
    for (int p = t; p < NPOS; p += 256) {
        int hh = p / 28, ww = p % 28;
        float z = 0.f;
        #pragma unroll
        for (int kh = 0; kh < 7; ++kh) {
            int ih = hh + kh - 3;
            if (ih < 0 || ih >= 28) continue;
            #pragma unroll
            for (int kw = 0; kw < 7; ++kw) {
                int iw = ww + kw - 3;
                if (iw < 0 || iw >= 28) continue;
                int q = ih * 28 + iw;
                z += sa_w[kh * 7 + kw]      * sAvg[q]
                   + sa_w[49 + kh * 7 + kw] * sMax[q];
            }
        }
        sAtt[p] = 1.0f / (1.0f + __expf(-z));
    }
    __syncthreads();

    // tiled transpose-pack to bf16 (both Xt and Qt)
    bf16* Xb = Xt + (size_t)b * NPOS * CH;
    bf16* Qb = Qt + (size_t)b * NPOS * CH;
    for (int c0 = 0; c0 < CH; c0 += 32) {
        for (int p0 = 0; p0 < NPOS; p0 += 64) {
            // load phase: lanes across positions (coalesced 128B/wave)
            for (int idx = t; idx < 32 * 64; idx += 256) {
                int cl = idx >> 6, pl = idx & 63;
                int p = p0 + pl;
                if (p < NPOS)
                    tile[cl][pl] = xb[(size_t)(c0 + cl) * NPOS + p];
            }
            __syncthreads();
            // store phase: lanes across channels (contiguous 64B/wave bf16)
            for (int idx = t; idx < 64 * 32; idx += 256) {
                int pl = idx >> 5, cl = idx & 31;
                int p = p0 + pl;
                if (p < NPOS) {
                    float v = tile[cl][pl];
                    size_t o = (size_t)p * CH + c0 + cl;
                    Xb[o] = (bf16)v;
                    Qb[o] = (bf16)(v * sAtt[p]);
                }
            }
            __syncthreads();
        }
    }
}

// ---------------------------------------------------------------------------
// Kernel 2: K projection. Kt[b][m][co] = key_w[co][:] . x[:][m] + key_b[co].
// Grid (49, 4, B); 8 waves/block, each wave owns one 16x16 output tile,
// K-loop over 512 channels in steps of 32 via v_wmma_f32_16x16x32_bf16.
// ---------------------------------------------------------------------------
__global__ void kgemm_kernel(const bf16* __restrict__ Xt,
                             const bf16* __restrict__ kwb,
                             const float* __restrict__ key_b,
                             bf16* __restrict__ Kt) {
    const int mt = blockIdx.x;                 // position tile
    const int b  = blockIdx.z;
    const int t = threadIdx.x, lane = t & 31, wave = t >> 5;
    const int hi = (lane >> 4) & 1, l16 = lane & 15;
    const int ct = blockIdx.y * 8 + wave;      // output-channel tile 0..31

    const bf16* arow = Xt + ((size_t)b * NPOS + (size_t)(mt * 16 + l16)) * CH;
    const bf16* brow = kwb + (size_t)(ct * 16 + l16) * CH;

    v8f acc = {};
    #pragma unroll 4
    for (int c0 = 0; c0 < CH; c0 += 32) {
        v16bf a  = load_frag(arow + c0 + hi * 8,  arow + c0 + hi * 8 + 16);
        v16bf bb = load_frag(brow + c0 + hi * 8,  brow + c0 + hi * 8 + 16);
        acc = __builtin_amdgcn_wmma_f32_16x16x32_bf16(
                  false, a, false, bb, (short)0, acc, false, false);
    }

    const float bias = key_b[ct * 16 + l16];
    bf16* Kb = Kt + (size_t)b * NPOS * CH;
    #pragma unroll
    for (int v = 0; v < 8; ++v) {
        int m = mt * 16 + v + hi * 8;          // C/D layout: row = v + 8*hi
        Kb[(size_t)m * CH + ct * 16 + l16] = (bf16)(acc[v] + bias);
    }
}

// ---------------------------------------------------------------------------
// Kernel 3: energy tile rows + softmax + per-tile column sums.
// Grid (49, B). Each block owns 16 energy rows x all 784 columns.
// 49 column tiles spread over 8 waves (7 accumulators each); S staged in LDS,
// then row max / row sum / column-sum-of-softmax. Deterministic (no atomics):
// per-tile partials go to s_part[b][nt][m].
// ---------------------------------------------------------------------------
__global__ void attn_kernel(const bf16* __restrict__ Qt,
                            const bf16* __restrict__ Kt,
                            float* __restrict__ s_part) {
    __shared__ float e[16][800];               // 16 rows x 784 (padded)
    __shared__ float red[16][16];
    __shared__ float rowMax[16], rowInv[16];

    const int nt = blockIdx.x, b = blockIdx.y;
    const int t = threadIdx.x, lane = t & 31, wave = t >> 5;
    const int hi = (lane >> 4) & 1, l16 = lane & 15;

    const bf16* Qb = Qt + (size_t)b * NPOS * CH;
    const bf16* Kb = Kt + (size_t)b * NPOS * CH;
    const bf16* arow = Qb + (size_t)(nt * 16 + l16) * CH;

    v8f acc[7] = {};
    for (int c0 = 0; c0 < CH; c0 += 32) {
        v16bf a = load_frag(arow + c0 + hi * 8, arow + c0 + hi * 8 + 16);
        #pragma unroll
        for (int j = 0; j < 7; ++j) {
            int mt = wave + 8 * j;             // wave-uniform predicate: EXEC stays full
            if (mt < NT) {
                const bf16* brow = Kb + (size_t)(mt * 16 + l16) * CH + c0 + hi * 8;
                v16bf bb = load_frag(brow, brow + 16);
                acc[j] = __builtin_amdgcn_wmma_f32_16x16x32_bf16(
                             false, a, false, bb, (short)0, acc[j], false, false);
            }
        }
    }

    // spill energy tiles to LDS: row n_local = v + 8*hi, col m = mt*16 + l16
    #pragma unroll
    for (int j = 0; j < 7; ++j) {
        int mt = wave + 8 * j;
        if (mt < NT) {
            #pragma unroll
            for (int v = 0; v < 8; ++v)
                e[v + hi * 8][mt * 16 + l16] = acc[j][v];
        }
    }
    __syncthreads();

    // row max (16 threads per row)
    const int r = t >> 4, k = t & 15;
    float pm = -3.4e38f;
    for (int m = k; m < NPOS; m += 16) pm = fmaxf(pm, e[r][m]);
    red[r][k] = pm;
    __syncthreads();
    if (k == 0) {
        float mm = red[r][0];
        #pragma unroll
        for (int i = 1; i < 16; ++i) mm = fmaxf(mm, red[r][i]);
        rowMax[r] = mm;
    }
    __syncthreads();

    // row sum of exp
    float ps = 0.f;
    {
        float mm = rowMax[r];
        for (int m = k; m < NPOS; m += 16) ps += __expf(e[r][m] - mm);
    }
    red[r][k] = ps;
    __syncthreads();
    if (k == 0) {
        float ss = 0.f;
        #pragma unroll
        for (int i = 0; i < 16; ++i) ss += red[r][i];
        rowInv[r] = 1.0f / ss;
    }
    __syncthreads();

    // column partial sums of softmax over this row tile
    float* sp = s_part + ((size_t)b * NT + nt) * NPOS;
    for (int m = t; m < NPOS; m += 256) {
        float a = 0.f;
        #pragma unroll
        for (int rr = 0; rr < 16; ++rr)
            a += __expf(e[rr][m] - rowMax[rr]) * rowInv[rr];
        sp[m] = a;
    }
}

// ---------------------------------------------------------------------------
// Kernel 4 (per batch): s[m] = sum_nt s_part; y[ci] = sum_m s[m]*x[ci][m]
// (and, fused in the same contiguous pass over x, mean_n x[b,ci,n]);
// out[c] = gamma*(value_w[c].y + value_b[c]*sum(s))/N + mean_x[c].
// ---------------------------------------------------------------------------
__global__ void final_kernel(const float* __restrict__ x,
                             const float* __restrict__ value_w,
                             const float* __restrict__ value_b,
                             const float* __restrict__ gamma,
                             const float* __restrict__ s_part,
                             float* __restrict__ out) {
    __shared__ float sS[NPOS];
    __shared__ float sY[CH];
    __shared__ float sMean[CH];
    __shared__ float sSum;
    const int b = blockIdx.x, t = threadIdx.x;

    for (int p = t; p < NPOS; p += 256) {
        float a = 0.f;
        for (int nt = 0; nt < NT; ++nt)
            a += s_part[((size_t)b * NT + nt) * NPOS + p];
        sS[p] = a;
    }
    __syncthreads();
    if (t == 0) {
        float a = 0.f;
        for (int p = 0; p < NPOS; ++p) a += sS[p];
        sSum = a;                                  // analytically == N
    }
    for (int ci = t; ci < CH; ci += 256) {
        const float* row = x + ((size_t)b * CH + ci) * NPOS;
        float a = 0.f, ms = 0.f;
        for (int p = 0; p < NPOS; ++p) {
            float v = row[p];
            a  += v * sS[p];
            ms += v;
        }
        sY[ci]    = a;
        sMean[ci] = ms * (1.0f / NPOS);
    }
    __syncthreads();

    const float g = gamma[0];
    for (int c = t; c < CH; c += 256) {
        const float* wrow = value_w + (size_t)c * CH;
        float a = 0.f;
        for (int ci = 0; ci < CH; ++ci) a += wrow[ci] * sY[ci];
        a += value_b[c] * sSum;
        out[b * CH + c] = g * a * (1.0f / NPOS) + sMean[c];
    }
}

// ---------------------------------------------------------------------------
extern "C" void kernel_launch(void* const* d_in, const int* in_sizes, int n_in,
                              void* d_out, int out_size, void* d_ws, size_t ws_size,
                              hipStream_t stream) {
    const float* x       = (const float*)d_in[0];
    const float* sa_w    = (const float*)d_in[1];
    const float* key_w   = (const float*)d_in[2];
    const float* key_b   = (const float*)d_in[3];
    const float* value_w = (const float*)d_in[4];
    const float* value_b = (const float*)d_in[5];
    const float* gamma   = (const float*)d_in[6];
    float* out = (float*)d_out;

    // Workspace carve-up (~165 MB; all chunks 16B aligned).
    const size_t szT = (size_t)BATCH * NPOS * CH * sizeof(bf16);   // 51,380,224 B
    char* p = (char*)d_ws;
    bf16*  Xt     = (bf16*)p;  p += szT;
    bf16*  Qt     = (bf16*)p;  p += szT;
    bf16*  Kt     = (bf16*)p;  p += szT;
    bf16*  kwb    = (bf16*)p;  p += (size_t)CH * CH * sizeof(bf16);
    float* s_part = (float*)p; p += (size_t)BATCH * NT * NPOS * sizeof(float);

    convert_w_kernel<<<dim3((CH * CH + 255) / 256), dim3(256), 0, stream>>>(key_w, kwb);
    prep_kernel<<<dim3(BATCH), dim3(256), 0, stream>>>(x, sa_w, Xt, Qt);
    kgemm_kernel<<<dim3(NT, 4, BATCH), dim3(256), 0, stream>>>(Xt, kwb, key_b, Kt);
    attn_kernel<<<dim3(NT, BATCH), dim3(256), 0, stream>>>(Qt, Kt, s_part);
    final_kernel<<<dim3(BATCH), dim3(256), 0, stream>>>(x, value_w, value_b, gamma,
                                                        s_part, out);
}
